// GroupedQuerySelfAttention_21148418966100
// MI455X (gfx1250) — matrix operand order repaired and checked by reading, not compiled, not run on hardware
//
#include <hip/hip_runtime.h>
#include <hip/hip_bf16.h>

// ---------------------------------------------------------------------------
// GQA self-attention for MI455X (gfx1250, wave32, WMMA).
// Pipeline: cvt x->bf16 | cvt+transpose W | GEMM(q,kv) | flash-attn | GEMM(out)
// All matmuls use v_wmma_f32_16x16x32_bf16 (f32 accumulate).
// GEMM waves compute 32x64 tiles (two M-subtiles share each B fragment):
// 12 b128 loads per 8 WMMAs instead of 10 per 4.
// Workspace assumption: ws_size >= 60 MiB.
// ---------------------------------------------------------------------------

#define BATCH 2
#define NTOK  2048
#define DMODEL 2048
#define HKV   8
#define GRP   4
#define CDIM  64
#define MTOT  (BATCH * NTOK)   // 4096 rows in all GEMMs

typedef __attribute__((ext_vector_type(16))) __bf16 v16bf;
typedef __attribute__((ext_vector_type(8)))  float  v8f;

struct __attribute__((aligned(16))) U4 { unsigned int x, y, z, w; };
struct __attribute__((aligned(16))) U8 { U4 lo, hi; };

__device__ __forceinline__ unsigned short f2bf(float f) {
  unsigned int u = __builtin_bit_cast(unsigned int, f);
  u += 0x7FFFu + ((u >> 16) & 1u);               // round-to-nearest-even
  return (unsigned short)(u >> 16);
}

// A-operand fragment (16x32 bf16, MxK): per-lane 16 contiguous K values {b..b+15}
__device__ __forceinline__ v16bf load_afrag(const unsigned short* p) {
  U8 t; t.lo = *(const U4*)p; t.hi = *(const U4*)(p + 8);
  return __builtin_bit_cast(v16bf, t);
}
// B-operand fragment (32x16 bf16, KxN): per-lane 16 contiguous K values
__device__ __forceinline__ v16bf load_bfrag(const unsigned short* p) {
  U8 t; t.lo = *(const U4*)p; t.hi = *(const U4*)(p + 8);
  return __builtin_bit_cast(v16bf, t);
}
__device__ __forceinline__ v8f wmma_bf16(v16bf a, v16bf b, v8f c) {
  return __builtin_amdgcn_wmma_f32_16x16x32_bf16(false, a, false, b, (short)0, c,
                                                 false, false);
}

// ---------------------------------------------------------------------------
__global__ __launch_bounds__(256) void k_cvt_bf16(const float* __restrict__ src,
                                                  unsigned short* __restrict__ dst,
                                                  int n) {
  for (int i = blockIdx.x * blockDim.x + threadIdx.x; i < n;
       i += gridDim.x * blockDim.x)
    dst[i] = f2bf(src[i]);
}

// src [rows][cols] f32 -> dst [cols][rows] bf16 (weights stored out-major)
__global__ __launch_bounds__(256) void k_cvt_transpose(const float* __restrict__ src,
                                                       unsigned short* __restrict__ dst,
                                                       int rows, int cols) {
  int total = rows * cols;
  for (int i = blockIdx.x * blockDim.x + threadIdx.x; i < total;
       i += gridDim.x * blockDim.x) {
    int o = i / rows, r = i - o * rows;
    dst[i] = f2bf(src[(size_t)r * cols + o]);
  }
}

// ---------------------------------------------------------------------------
// Y = A[4096xK] * Bt[NoutxK]^T ; one wave -> 32 rows x 64 cols (2 M-subtiles).
// mode 0: q projection (scale 1/sqrt(32), scatter to qb[b][h][g][n][c])
// mode 1: kv projection (k -> kb[b][h][n][c], v -> vtb[b][h][c][n])
// mode 2: output projection (+bias, f32 to outp)
__global__ __launch_bounds__(256) void k_gemm_bf16(
    const unsigned short* __restrict__ A, const unsigned short* __restrict__ Bt,
    int K, int Nout, int mode,
    unsigned short* __restrict__ qb, unsigned short* __restrict__ kb,
    unsigned short* __restrict__ vtb,
    float* __restrict__ outp, const float* __restrict__ bias) {
  const int lane   = threadIdx.x & 31;
  const int wv     = threadIdx.x >> 5;
  const int waveId = blockIdx.x * (blockDim.x >> 5) + wv;
  const int nBlk   = Nout >> 6;
  if (waveId >= (MTOT / 32) * nBlk) return;
  const int mTile = waveId / nBlk;               // 32-row block
  const int n0    = (waveId - mTile * nBlk) << 6;
  const int col   = lane & 15;
  const int abase = (lane < 16) ? 0 : 16;
  const int bbase = (lane < 16) ? 0 : 16;

  const v8f zf = {0.f, 0.f, 0.f, 0.f, 0.f, 0.f, 0.f, 0.f};
  v8f acc[2][4];
#pragma unroll
  for (int sm = 0; sm < 2; ++sm)
#pragma unroll
    for (int t = 0; t < 4; ++t) acc[sm][t] = zf;

  const unsigned short* arow0 = A + (size_t)(mTile * 32 + col) * K;
  const unsigned short* arow1 = arow0 + (size_t)16 * K;
  for (int k0 = 0; k0 < K; k0 += 32) {
    v16bf af0 = load_afrag(arow0 + k0 + abase);
    v16bf af1 = load_afrag(arow1 + k0 + abase);
    __builtin_prefetch(arow0 + k0 + 64, 0, 3);   // global_prefetch_b8, near scope
    __builtin_prefetch(arow1 + k0 + 64, 0, 3);
#pragma unroll
    for (int t = 0; t < 4; ++t) {
      const unsigned short* brow =
          Bt + (size_t)(n0 + t * 16 + col) * K + k0 + bbase;
      v16bf bf = load_bfrag(brow);
      acc[0][t] = wmma_bf16(af0, bf, acc[0][t]);
      acc[1][t] = wmma_bf16(af1, bf, acc[1][t]);
    }
  }

#pragma unroll
  for (int sm = 0; sm < 2; ++sm) {
#pragma unroll
    for (int t = 0; t < 4; ++t) {
#pragma unroll
      for (int j = 0; j < 8; ++j) {
        int row = mTile * 32 + sm * 16 + ((lane < 16) ? j : j + 8); // token B*N
        int jc  = n0 + t * 16 + col;                               // out channel
        float v = acc[sm][t][j];
        int bb = row >> 11, nn = row & (NTOK - 1);
        if (mode == 0) {
          v *= 0.17677669529663689f;                     // 1/sqrt(HQ)=1/sqrt(32)
          int h = jc >> 8, g = (jc >> 6) & 3, c = jc & 63;
          qb[((((size_t)(bb * HKV + h) * GRP + g) * NTOK + nn) << 6) + c] =
              f2bf(v);
        } else if (mode == 1) {
          if (jc < HKV * CDIM) {
            int h = jc >> 6, c = jc & 63;
            kb[(((size_t)(bb * HKV + h) * NTOK + nn) << 6) + c] = f2bf(v);
          } else {
            int j2 = jc - HKV * CDIM;
            int h = j2 >> 6, c = j2 & 63;
            vtb[((size_t)(bb * HKV + h) * CDIM + c) * NTOK + nn] = f2bf(v);
          }
        } else {
          outp[(size_t)row * DMODEL + jc] = v + bias[jc];
        }
      }
    }
  }
}

// ---------------------------------------------------------------------------
// Flash attention: one wave = 16 query rows of one (b,h,g). s-step = 32.
__global__ __launch_bounds__(256) void k_attn(
    const unsigned short* __restrict__ qb, const unsigned short* __restrict__ kb,
    const unsigned short* __restrict__ vtb, unsigned short* __restrict__ ob) {
  __shared__ __attribute__((aligned(16))) unsigned short plds[8][512]; // 16x32/wave

  const int lane = threadIdx.x & 31;
  const int wv   = threadIdx.x >> 5;
  const int w    = blockIdx.x * 8 + wv;
  const int mTile = w & 127;
  const int g     = (w >> 7) & 3;
  const int h     = (w >> 9) & 7;
  const int b     = w >> 12;
  const int col   = lane & 15;
  const int abase = (lane < 16) ? 0 : 16;
  const int bbase = (lane < 16) ? 0 : 16;

  // Q fragments for this wave's 16 rows (C = 64 -> two K=32 A-frags)
  const unsigned short* qrow =
      qb + ((((size_t)(b * HKV + h) * GRP + g) * NTOK) + mTile * 16 + col) * CDIM;
  v16bf aq0 = load_afrag(qrow + abase);
  v16bf aq1 = load_afrag(qrow + 32 + abase);

  const unsigned short* kbase = kb + (size_t)(b * HKV + h) * NTOK * CDIM;
  const unsigned short* vbase = vtb + (size_t)(b * HKV + h) * CDIM * NTOK;

  const v8f zf = {0.f, 0.f, 0.f, 0.f, 0.f, 0.f, 0.f, 0.f};
  float mi[8], li[8];
  v8f oacc[4];
#pragma unroll
  for (int j = 0; j < 8; ++j) { mi[j] = -3.0e38f; li[j] = 0.f; }
#pragma unroll
  for (int t = 0; t < 4; ++t) oacc[t] = zf;

  for (int s0 = 0; s0 < NTOK; s0 += 32) {
    // --- S = Q K^T for 32 key columns (two 16x16 tiles) ---
    const unsigned short* krow0 = kbase + (size_t)(s0 + col) * CDIM;
    const unsigned short* krow1 = kbase + (size_t)(s0 + 16 + col) * CDIM;
    __builtin_prefetch(krow0 + 32 * CDIM, 0, 3);     // next s-tile K rows
    v8f sA = wmma_bf16(aq0, load_bfrag(krow0 + bbase), zf);
    sA     = wmma_bf16(aq1, load_bfrag(krow0 + 32 + bbase), sA);
    v8f sB = wmma_bf16(aq0, load_bfrag(krow1 + bbase), zf);
    sB     = wmma_bf16(aq1, load_bfrag(krow1 + 32 + bbase), sB);

    // --- online softmax (rows live in 16-lane halves per C/D layout) ---
#pragma unroll
    for (int j = 0; j < 8; ++j) {
      float x0 = sA[j], x1 = sB[j];
      float mrow = fmaxf(x0, x1);
      mrow = fmaxf(mrow, __shfl_xor(mrow, 1, 32));
      mrow = fmaxf(mrow, __shfl_xor(mrow, 2, 32));
      mrow = fmaxf(mrow, __shfl_xor(mrow, 4, 32));
      mrow = fmaxf(mrow, __shfl_xor(mrow, 8, 32));
      float mnew = fmaxf(mi[j], mrow);
      float p0 = __expf(x0 - mnew);
      float p1 = __expf(x1 - mnew);
      float corr = __expf(mi[j] - mnew);
      float ps = p0 + p1;
      ps += __shfl_xor(ps, 1, 32);
      ps += __shfl_xor(ps, 2, 32);
      ps += __shfl_xor(ps, 4, 32);
      ps += __shfl_xor(ps, 8, 32);
      li[j] = li[j] * corr + ps;
      mi[j] = mnew;
      oacc[0][j] *= corr; oacc[1][j] *= corr;
      oacc[2][j] *= corr; oacc[3][j] *= corr;
      int row = (lane < 16) ? j : j + 8;
      plds[wv][row * 32 + col]      = f2bf(p0);
      plds[wv][row * 32 + 16 + col] = f2bf(p1);
    }
    asm volatile("s_wait_dscnt 0" ::: "memory");   // P tile visible within wave

    // --- re-fetch P as A-fragment (16x32), then O += P V ---
    v16bf pf = load_afrag(&plds[wv][col * 32 + abase]);
#pragma unroll
    for (int t = 0; t < 4; ++t) {
      const unsigned short* vrow =
          vbase + (size_t)(t * 16 + col) * NTOK + s0 + bbase;
      __builtin_prefetch(vrow + 32, 0, 3);           // next s-tile V chunk
      oacc[t] = wmma_bf16(pf, load_bfrag(vrow), oacc[t]);
    }
  }

  // --- finalize: O /= l, scatter to ob[b][n][(h g c)] ---
  float inv[8];
#pragma unroll
  for (int j = 0; j < 8; ++j) inv[j] = 1.0f / li[j];
#pragma unroll
  for (int t = 0; t < 4; ++t) {
#pragma unroll
    for (int j = 0; j < 8; ++j) {
      int token = mTile * 16 + ((lane < 16) ? j : j + 8);
      int chan  = h * (GRP * CDIM) + g * CDIM + t * 16 + col;
      ob[((size_t)b * NTOK + token) * DMODEL + chan] = f2bf(oacc[t][j] * inv[j]);
    }
  }
}

// ---------------------------------------------------------------------------
extern "C" void kernel_launch(void* const* d_in, const int* in_sizes, int n_in,
                              void* d_out, int out_size, void* d_ws, size_t ws_size,
                              hipStream_t stream) {
  const float* x   = (const float*)d_in[0];
  const float* Wq  = (const float*)d_in[1];
  const float* Wkv = (const float*)d_in[2];
  const float* Wp  = (const float*)d_in[3];
  const float* bp  = (const float*)d_in[4];
  float* out = (float*)d_out;

  char* ws = (char*)d_ws;
  const size_t MiB = 1024 * 1024;
  unsigned short* xb   = (unsigned short*)(ws + 0 * MiB);   // 16 MiB; reused as ob
  unsigned short* Wqt  = (unsigned short*)(ws + 16 * MiB);  // 8 MiB
  unsigned short* Wkvt = (unsigned short*)(ws + 24 * MiB);  // 4 MiB
  unsigned short* Wpt  = (unsigned short*)(ws + 28 * MiB);  // 8 MiB
  unsigned short* qbuf = (unsigned short*)(ws + 36 * MiB);  // 16 MiB
  unsigned short* kbuf = (unsigned short*)(ws + 52 * MiB);  // 4 MiB
  unsigned short* vtb  = (unsigned short*)(ws + 56 * MiB);  // 4 MiB -> 60 MiB total
  unsigned short* ob   = xb;  // x is dead once projections finish

  // precision conversions / weight transposes
  k_cvt_bf16<<<2048, 256, 0, stream>>>(x, xb, MTOT * DMODEL);
  k_cvt_transpose<<<2048, 256, 0, stream>>>(Wq, Wqt, DMODEL, DMODEL);
  k_cvt_transpose<<<1024, 256, 0, stream>>>(Wkv, Wkvt, DMODEL, 2 * HKV * CDIM);
  k_cvt_transpose<<<2048, 256, 0, stream>>>(Wp, Wpt, DMODEL, DMODEL);

  // q = (x Wq)/sqrt(32): 128 m-blocks * 32 n-blocks = 4096 waves -> 512 blocks
  k_gemm_bf16<<<512, 256, 0, stream>>>(xb, Wqt, DMODEL, DMODEL, 0,
                                       qbuf, kbuf, vtb, out, bp);
  // kv = x Wkv: 128 * 16 = 2048 waves -> 256 blocks
  k_gemm_bf16<<<256, 256, 0, stream>>>(xb, Wkvt, DMODEL, 2 * HKV * CDIM, 1,
                                       qbuf, kbuf, vtb, out, bp);
  // attention: B*H*G*(N/16) = 8192 waves -> 1024 blocks
  k_attn<<<1024, 256, 0, stream>>>(qbuf, kbuf, vtb, ob);
  // out = ob Wp + bp
  k_gemm_bf16<<<512, 256, 0, stream>>>(ob, Wpt, DMODEL, DMODEL, 2,
                                       qbuf, kbuf, vtb, out, bp);
}
